// FullyDynamicSTGNN_31061203485227
// MI455X (gfx1250) — compile-verified, matrix operand order checked by reading, not run on hardware
//
#include <hip/hip_runtime.h>
#include <hip/hip_bf16.h>

typedef _Float16 half_t;
typedef __attribute__((ext_vector_type(16))) _Float16 v16h;
typedef __attribute__((ext_vector_type(8)))  _Float16 v8h;
typedef __attribute__((ext_vector_type(8)))  float    v8f;

#define B_   32
#define T_   64
#define N_   128
#define FIN  16
#define H_   64
#define SOUT 16
#define D_   2048
#define NHEADS 16
#define DH   128
#define DFF  2048

__device__ __forceinline__ v8f vzero8() { v8f z = {}; return z; }

__device__ __forceinline__ v8f wmma_f16(v16h a, v16h b, v8f c) {
  // D = A(16x32 f16) x B(32x16 f16) + C(16x16 f32)
  return __builtin_amdgcn_wmma_f32_16x16x32_f16(
      /*neg_a=*/false, a, /*neg_b=*/false, b,
      /*c_mod=*/(short)0, c, /*reuse_a=*/false, /*reuse_b=*/false);
}

// ---- CDNA5 async global->LDS copy (ASYNCcnt-tracked, bypasses VGPRs) -------
__device__ __forceinline__ unsigned lds_off32(const void* p) {
  // generic LDS address: low 32 bits are the LDS byte offset (ISA flat->LDS rule)
  return (unsigned)(unsigned long long)p;
}
__device__ __forceinline__ void async_copy16(unsigned lds_off, const half_t* g) {
  asm volatile("global_load_async_to_lds_b128 %0, %1, off"
               :: "v"(lds_off), "v"(g) : "memory");
}
__device__ __forceinline__ void wait_async() {
  asm volatile("s_wait_asynccnt 0x0" ::: "memory");
}

// Row-contiguous fragment load: p = row base at current 32-wide k window.
// lane&15 selects the row/col; kbase = (lane<16)?0:8 per ISA 16-bit A/B layout.
__device__ __forceinline__ v16h ld_frag_row(const half_t* p, int kbase) {
  union { v16h v; v8h h[2]; } u;
  u.h[0] = *(const v8h*)(p + kbase);       // k = kbase..kbase+7
  u.h[1] = *(const v8h*)(p + 16 + kbase);  // k = 16+kbase..16+kbase+7
  return u.v;
}

// Strided (column) fragment load: element j -> k = kbase+j (j<8) else 8+kbase+j
__device__ __forceinline__ v16h ld_frag_strided(const half_t* p, int stride, int kbase) {
  v16h f;
#pragma unroll
  for (int j = 0; j < 16; ++j) {
    int k = (j < 8) ? (kbase + j) : (8 + kbase + j);
    f[j] = p[k * stride];
  }
  return f;
}

// ---------------------------------------------------------------- cvt f32->f16
__global__ __launch_bounds__(256) void cvt_f32_f16(const float* __restrict__ s,
                                                   half_t* __restrict__ d, int n) {
  int i = blockIdx.x * 256 + threadIdx.x;
  int stride = gridDim.x * 256;
  for (; i < n; i += stride) d[i] = (half_t)s[i];
}

// ---------------------------------------------------------------- GEMM (WMMA)
// C[M,N] = A[M,K](f16,row) x B[K,N](f16,row) + bias ; optional relu
// Tile: BM=128, BN=128, BK=32 ; 256 threads = 8 waves (4x2), wave = 32x64.
#define BM 128
#define BN 128
#define BK 32
#define LDA 40
#define LDB 40

__global__ __launch_bounds__(256) void gemm_wmma_kernel(
    const half_t* __restrict__ A, const half_t* __restrict__ Bm,
    const float* __restrict__ bias, float* __restrict__ Cf,
    half_t* __restrict__ Ch, int M, int N, int K, int doRelu)
{
  __shared__ alignas(16) half_t As[2][BM * LDA];
  __shared__ alignas(16) half_t Bs[2][BN * LDB];

  const int tid  = threadIdx.x;
  const int lane = tid & 31;
  const int wv   = tid >> 5;
  const int l16  = lane & 15, hi = lane >> 4, kbase = hi ? 8 : 0;
  const int wm = wv >> 1, wn = wv & 1;
  const int bm0 = blockIdx.y * BM, bn0 = blockIdx.x * BN;
  const int nk = K / BK;

  const int arow = tid >> 1, aseg = tid & 1;   // A loader: 128 rows x 2 segs
  const int brow = tid >> 3, bseg = tid & 7;   // B loader: 32 k-rows x 8 segs

  auto load_tile = [&](int s, int kt) {
    // A tile: async global->LDS (two b128 per thread, no VGPR round-trip)
    const half_t* ga = A + (size_t)(bm0 + arow) * K + kt * BK + aseg * 16;
    unsigned la = lds_off32(&As[s][arow * LDA + aseg * 16]);
    async_copy16(la, ga);
    async_copy16(la + 16, ga + 8);
    // B tile: transpose into Bs[n][k] via VGPRs
    const half_t* gb = Bm + (size_t)(kt * BK + brow) * N + bn0 + bseg * 16;
    v8h b0 = *(const v8h*)(gb);
    v8h b1 = *(const v8h*)(gb + 8);
#pragma unroll
    for (int j = 0; j < 8; ++j) Bs[s][(bseg * 16 + j) * LDB + brow] = b0[j];
#pragma unroll
    for (int j = 0; j < 8; ++j) Bs[s][(bseg * 16 + 8 + j) * LDB + brow] = b1[j];
  };

  v8f acc[2][4];
#pragma unroll
  for (int i = 0; i < 2; ++i)
#pragma unroll
    for (int j = 0; j < 4; ++j) acc[i][j] = vzero8();

  load_tile(0, 0);
  for (int kt = 0; kt < nk; ++kt) {
    wait_async();        // ASYNCcnt not compiler-tracked: drain before barrier
    __syncthreads();
    if (kt + 1 < nk) load_tile((kt + 1) & 1, kt + 1);
    if (kt + 2 < nk) {
      __builtin_prefetch(A + (size_t)(bm0 + arow) * K + (kt + 2) * BK + aseg * 16, 0, 1);
      __builtin_prefetch(Bm + (size_t)((kt + 2) * BK + brow) * N + bn0 + bseg * 16, 0, 1);
    }
    const int s = kt & 1;
    v16h afr[2], bfr[4];
#pragma unroll
    for (int i = 0; i < 2; ++i)
      afr[i] = ld_frag_row(&As[s][(wm * 32 + i * 16 + l16) * LDA], kbase);
#pragma unroll
    for (int j = 0; j < 4; ++j)
      bfr[j] = ld_frag_row(&Bs[s][(wn * 64 + j * 16 + l16) * LDB], kbase);
#pragma unroll
    for (int i = 0; i < 2; ++i)
#pragma unroll
      for (int j = 0; j < 4; ++j)
        acc[i][j] = wmma_f16(afr[i], bfr[j], acc[i][j]);
  }

#pragma unroll
  for (int i = 0; i < 2; ++i)
#pragma unroll
    for (int j = 0; j < 4; ++j)
#pragma unroll
      for (int g = 0; g < 8; ++g) {
        int m = bm0 + wm * 32 + i * 16 + g + hi * 8;
        int n = bn0 + wn * 64 + j * 16 + l16;
        float v = acc[i][j][g];
        if (bias) v += bias[n];
        if (doRelu) v = v > 0.f ? v : 0.f;
        if (Cf) Cf[(size_t)m * N + n] = v;
        if (Ch) Ch[(size_t)m * N + n] = (half_t)v;
      }
}

// ---------------------------------------------------------------- attention
// One block per (b, head). S = QK^T/sqrt(dh); softmax; O = P V. All WMMA.
#define A_LDQ 136
#define A_LDVT 72
#define A_LDSS 68
#define A_LDP 72

__global__ __launch_bounds__(256) void attn_kernel(
    const half_t* __restrict__ q, const half_t* __restrict__ k,
    const half_t* __restrict__ v, half_t* __restrict__ ao)
{
  __shared__ alignas(16) char smem[17408 + 18432 + 17408];
  half_t* Qs  = (half_t*)smem;                   // [64][136]
  half_t* Ks  = (half_t*)(smem + 17408);         // [64][136]
  half_t* VsT = (half_t*)(smem + 17408);         // [128][72] reuses Ks
  half_t* Ps  = (half_t*)smem;                   // [64][72]  reuses Qs
  float*  Ss  = (float*)(smem + 17408 + 18432);  // [64][68]

  const int tid = threadIdx.x;
  const int lane = tid & 31, wv = tid >> 5;
  const int l16 = lane & 15, hi = lane >> 4, kbase = hi ? 8 : 0;
  const int b = blockIdx.x / NHEADS, h = blockIdx.x % NHEADS;
  const size_t rowBase = (size_t)(b * T_) * D_ + (size_t)h * DH;

  { // stage Q, K (64 x 128 f16 each) via async global->LDS
    int tok = tid >> 2, seg = tid & 3;
    const half_t* gq = q + rowBase + (size_t)tok * D_ + seg * 32;
    const half_t* gk = k + rowBase + (size_t)tok * D_ + seg * 32;
    unsigned lq = lds_off32(Qs + tok * A_LDQ + seg * 32);
    unsigned lk = lds_off32(Ks + tok * A_LDQ + seg * 32);
#pragma unroll
    for (int c = 0; c < 4; ++c) {
      async_copy16(lq + c * 16, gq + c * 8);
      async_copy16(lk + c * 16, gk + c * 8);
    }
  }
  wait_async();
  __syncthreads();

  // S = Q K^T * scale  (M=64,N=64,K=128): 16 tiles over 8 waves (2 each)
  const int mt = wv >> 1;
  const int ntb = (wv & 1) * 2;
  v8f sacc[2]; sacc[0] = vzero8(); sacc[1] = vzero8();
  for (int kk = 0; kk < 4; ++kk) {
    v16h af = ld_frag_row(&Qs[(mt * 16 + l16) * A_LDQ + kk * 32], kbase);
#pragma unroll
    for (int j = 0; j < 2; ++j) {
      v16h bf = ld_frag_row(&Ks[((ntb + j) * 16 + l16) * A_LDQ + kk * 32], kbase);
      sacc[j] = wmma_f16(af, bf, sacc[j]);
    }
  }
  const float scale = 0.088388347648318f; // 1/sqrt(128)
#pragma unroll
  for (int j = 0; j < 2; ++j)
#pragma unroll
    for (int g = 0; g < 8; ++g) {
      int m = mt * 16 + g + hi * 8;
      int n = (ntb + j) * 16 + l16;
      Ss[m * A_LDSS + n] = sacc[j][g] * scale;
    }
  __syncthreads();

  { // stage V transposed (overwrites Ks region)
    int tok = tid >> 2, seg = tid & 3;
    const half_t* gv = v + rowBase + (size_t)tok * D_ + seg * 32;
#pragma unroll
    for (int j = 0; j < 32; ++j)
      VsT[(seg * 32 + j) * A_LDVT + tok] = gv[j];
  }
  if (tid < 64) { // softmax row tid -> P (overwrites Qs region)
    float mx = -1e30f;
    for (int n = 0; n < 64; ++n) mx = fmaxf(mx, Ss[tid * A_LDSS + n]);
    float sum = 0.f;
    for (int n = 0; n < 64; ++n) {
      float e = __expf(Ss[tid * A_LDSS + n] - mx);
      sum += e;
      Ps[tid * A_LDP + n] = (half_t)e;
    }
    float inv = 1.f / sum;
    for (int n = 0; n < 64; ++n)
      Ps[tid * A_LDP + n] = (half_t)((float)Ps[tid * A_LDP + n] * inv);
  }
  __syncthreads();

  // O = P V  (M=64,N=128,K=64): 32 tiles over 8 waves (4 each)
  const int omt = wv >> 1;
  const int onb = (wv & 1) * 4;
  v8f oacc[4];
#pragma unroll
  for (int j = 0; j < 4; ++j) oacc[j] = vzero8();
  for (int kk = 0; kk < 2; ++kk) {
    v16h af = ld_frag_row(&Ps[(omt * 16 + l16) * A_LDP + kk * 32], kbase);
#pragma unroll
    for (int j = 0; j < 4; ++j) {
      v16h bf = ld_frag_row(&VsT[((onb + j) * 16 + l16) * A_LDVT + kk * 32], kbase);
      oacc[j] = wmma_f16(af, bf, oacc[j]);
    }
  }
#pragma unroll
  for (int j = 0; j < 4; ++j)
#pragma unroll
    for (int g = 0; g < 8; ++g) {
      int m = omt * 16 + g + hi * 8;
      int n = (onb + j) * 16 + l16;
      ao[rowBase + (size_t)m * D_ + n] = (half_t)oacc[j][g];
    }
}

// ---------------------------------------------------------------- GAT
// One block per (t,b). Scores in VALU, aggregation + Wfc via WMMA.
#define G_ST 136   // xlT / xrS / GT stride (128 nodes + pad)
#define G_SA 136   // alphaT stride
#define G_SW 72    // WfcT stride

__global__ __launch_bounds__(256) void gat_kernel(
    const float* __restrict__ X, const float* __restrict__ V_Adap,
    const float* __restrict__ Wl, const float* __restrict__ bl,
    const float* __restrict__ Wr, const float* __restrict__ br,
    const float* __restrict__ att, const float* __restrict__ gbias,
    const float* __restrict__ Wfc, const float* __restrict__ bfc,
    const int* __restrict__ cls, float* __restrict__ hf32,
    half_t* __restrict__ hf16)
{
  __shared__ alignas(16) char smem[17408 + 34816 + 256 + 1024 + 1024 + 2304];
  half_t* xlT    = (half_t*)smem;                         // [64][136]; later GT
  half_t* alphaT = (half_t*)(smem + 17408);               // [128][136]; first xrS
  half_t* xrS    = alphaT;                                // [64][136]
  float*  attS   = (float*)(smem + 17408 + 34816);
  float*  redM   = (float*)(smem + 17408 + 34816 + 256);
  float*  redS   = (float*)(smem + 17408 + 34816 + 256 + 1024);
  half_t* WfcT   = (half_t*)(smem + 17408 + 34816 + 256 + 2048); // [16][72]

  const int tid = threadIdx.x;
  const int lane = tid & 31, wv = tid >> 5;
  const int l16 = lane & 15, hi = lane >> 4, kbase = hi ? 8 : 0;
  const int b  = blockIdx.x & (B_ - 1);
  const int tt = blockIdx.x >> 5;

  { // phase 1: xl = xW_l+b_l, xr = xW_r+b_r (transposed f16 in LDS)
    int n = tid >> 1, hh = (tid & 1) * 32;
    float xv[FIN];
    const float* xp = X + ((size_t)(b * T_ + tt)) * (N_ * FIN) + n * FIN;
#pragma unroll
    for (int f = 0; f < FIN; ++f) xv[f] = xp[f];
    for (int h = hh; h < hh + 32; ++h) {
      float al = bl[h], ar = br[h];
#pragma unroll
      for (int f = 0; f < FIN; ++f) {
        al += xv[f] * Wl[f * H_ + h];
        ar += xv[f] * Wr[f * H_ + h];
      }
      xlT[h * G_ST + n] = (half_t)al;
      xrS[h * G_ST + n] = (half_t)ar;
    }
    if (tid < H_) attS[tid] = att[tid];
#pragma unroll
    for (int u = 0; u < 4; ++u) {
      int idx = tid * 4 + u;           // 0..1023
      int o = idx & 15, kk2 = idx >> 4;
      WfcT[o * G_SW + kk2] = (half_t)Wfc[kk2 * SOUT + o];
    }
  }
  __syncthreads();

  { // phase 2: masked scores + softmax over sources -> alphaT[dst][src]
    const int d = tid >> 1;
    const int sb = (tid & 1) * 64;
    float xrv[H_];
#pragma unroll
    for (int h = 0; h < H_; ++h) xrv[h] = (float)xrS[h * G_ST + d];
    const int ci = cls[0];
    const float* va = V_Adap + (size_t)ci * N_ * N_;
    float ev[64];
    float mx = -1e30f;
    for (int si = 0; si < 64; ++si) {
      int s = sb + si;
      float acc = 0.f;
      for (int h = 0; h < H_; ++h) {
        float u = (float)xlT[h * G_ST + s] + xrv[h];
        u = (u > 0.f) ? u : 0.2f * u;
        acc += attS[h] * u;
      }
      float aval = (1.f / 128.f) / (1.f + __expf(-va[s * N_ + d]));
      bool keep = (aval > 0.004f) || (s == d);
      ev[si] = keep ? acc : -1e9f;
      mx = fmaxf(mx, ev[si]);
    }
    redM[d * 2 + (tid & 1)] = mx;
    __syncthreads();
    float M = fmaxf(redM[d * 2], redM[d * 2 + 1]);
    float sum = 0.f;
    for (int si = 0; si < 64; ++si) { ev[si] = __expf(ev[si] - M); sum += ev[si]; }
    redS[d * 2 + (tid & 1)] = sum;
    __syncthreads();
    float inv = 1.f / (redS[d * 2] + redS[d * 2 + 1]);
    for (int si = 0; si < 64; ++si)
      alphaT[d * G_SA + (sb + si)] = (half_t)(ev[si] * inv);
  }
  __syncthreads();

  // phase 3: out2 = alpha^T @ xl (M=128 dst, N=64 h, K=128 src), WMMA
  v8f g3[4];
#pragma unroll
  for (int j = 0; j < 4; ++j) g3[j] = vzero8();
  for (int kk = 0; kk < 4; ++kk) {
    v16h af = ld_frag_row(&alphaT[(wv * 16 + l16) * G_SA + kk * 32], kbase);
#pragma unroll
    for (int j = 0; j < 4; ++j) {
      v16h bf = ld_frag_row(&xlT[(j * 16 + l16) * G_ST + kk * 32], kbase);
      g3[j] = wmma_f16(af, bf, g3[j]);
    }
  }
  __syncthreads();
  // store GT[h][m] = out2[m][h] + gat_bias[h] into the (now free) xlT region
#pragma unroll
  for (int j = 0; j < 4; ++j)
#pragma unroll
    for (int g = 0; g < 8; ++g) {
      int m = wv * 16 + g + hi * 8;
      int n = j * 16 + l16;
      xlT[n * G_ST + m] = (half_t)(g3[j][g] + gbias[n]);
    }
  __syncthreads();

  // phase 4: out3 = G @ Wfc (M=128, N=16, K=64), WMMA
  v8f o4 = vzero8();
  for (int kk = 0; kk < 2; ++kk) {
    v16h af = ld_frag_strided(&xlT[(kk * 32) * G_ST + (wv * 16 + l16)], G_ST, kbase);
    v16h bf = ld_frag_row(&WfcT[l16 * G_SW + kk * 32], kbase);
    o4 = wmma_f16(af, bf, o4);
  }
  const size_t row = (size_t)(b * T_ + tt) * D_;
#pragma unroll
  for (int g = 0; g < 8; ++g) {
    int m = wv * 16 + g + hi * 8;
    int o = l16;
    float val = o4[g] + bfc[o];
    hf32[row + m * SOUT + o] = val;
    hf16[row + m * SOUT + o] = (half_t)val;
  }
}

// ---------------------------------------------------------------- residual+LN
__global__ __launch_bounds__(256) void add_ln_kernel(
    float* __restrict__ hf32, half_t* __restrict__ hf16,
    const float* __restrict__ delta, const float* __restrict__ g,
    const float* __restrict__ bta)
{
  __shared__ float red[256];
  const int row = blockIdx.x, tid = threadIdx.x;
  float* hrow = hf32 + (size_t)row * D_;
  const float* drow = delta + (size_t)row * D_;
  float s = 0.f;
  for (int i = tid; i < D_; i += 256) s += hrow[i] + drow[i];
  red[tid] = s; __syncthreads();
  for (int off = 128; off > 0; off >>= 1) {
    if (tid < off) red[tid] += red[tid + off];
    __syncthreads();
  }
  float mu = red[0] / (float)D_;
  __syncthreads();
  float s2 = 0.f;
  for (int i = tid; i < D_; i += 256) {
    float x = hrow[i] + drow[i] - mu;
    s2 += x * x;
  }
  red[tid] = s2; __syncthreads();
  for (int off = 128; off > 0; off >>= 1) {
    if (tid < off) red[tid] += red[tid + off];
    __syncthreads();
  }
  float inv = rsqrtf(red[0] / (float)D_ + 1e-5f);
  for (int i = tid; i < D_; i += 256) {
    float x = (hrow[i] + drow[i] - mu) * inv * g[i] + bta[i];
    hrow[i] = x;
    hf16[(size_t)row * D_ + i] = (half_t)x;
  }
}

// ---------------------------------------------------------------- final proj
__global__ __launch_bounds__(256) void final_kernel(
    const float* __restrict__ h, const float* __restrict__ Wout,
    const float* __restrict__ bout, float* __restrict__ out)
{
  int t = blockIdx.x * 256 + threadIdx.x;
  if (t >= B_ * 10) return;
  int b = t / 10, o = t % 10;
  const float* row = h + ((size_t)(b * T_ + (T_ - 1))) * D_;
  float acc = bout[o];
  for (int d = 0; d < D_; ++d) acc += row[d] * Wout[d * 10 + o];
  out[t] = acc;
}

// ---------------------------------------------------------------- launcher
extern "C" void kernel_launch(void* const* d_in, const int* in_sizes, int n_in,
                              void* d_out, int out_size, void* d_ws, size_t ws_size,
                              hipStream_t stream) {
  (void)in_sizes; (void)n_in; (void)out_size; (void)ws_size;
  const float* X      = (const float*)d_in[0];
  const float* V_Adap = (const float*)d_in[1];
  const float* Wl  = (const float*)d_in[2];
  const float* bl  = (const float*)d_in[3];
  const float* Wr  = (const float*)d_in[4];
  const float* br  = (const float*)d_in[5];
  const float* att = (const float*)d_in[6];
  const float* gb  = (const float*)d_in[7];
  const float* Wfc = (const float*)d_in[8];
  const float* bfc = (const float*)d_in[9];
  const float* Wq  = (const float*)d_in[10];
  const float* bq  = (const float*)d_in[11];
  const float* Wk  = (const float*)d_in[12];
  const float* bk  = (const float*)d_in[13];
  const float* Wv  = (const float*)d_in[14];
  const float* bv  = (const float*)d_in[15];
  const float* Wo  = (const float*)d_in[16];
  const float* bo  = (const float*)d_in[17];
  const float* ln1g = (const float*)d_in[18];
  const float* ln1b = (const float*)d_in[19];
  const float* W1  = (const float*)d_in[20];
  const float* b1  = (const float*)d_in[21];
  const float* W2  = (const float*)d_in[22];
  const float* b2  = (const float*)d_in[23];
  const float* ln2g = (const float*)d_in[24];
  const float* ln2b = (const float*)d_in[25];
  const float* Wout = (const float*)d_in[26];
  const float* bout = (const float*)d_in[27];
  const int*   cls  = (const int*)d_in[28];
  float* out = (float*)d_out;

  char* ws = (char*)d_ws;
  const size_t TOKD = (size_t)B_ * T_ * D_;   // 4M elements
  float*  hf32 = (float*)(ws);                       // 16 MB
  half_t* hf16 = (half_t*)(ws + TOKD * 4);           //  8 MB
  float*  tmpf = (float*)(ws + TOKD * 6);            // 16 MB
  half_t* qf   = (half_t*)(ws + TOKD * 10);          //  8 MB
  half_t* kf   = (half_t*)(ws + TOKD * 12);          //  8 MB
  half_t* vf   = (half_t*)(ws + TOKD * 14);          //  8 MB
  half_t* xf   = (half_t*)(ws + TOKD * 16);          //  8 MB (ao / ff1)
  half_t* wf   = (half_t*)(ws + TOKD * 18);          //  8 MB (f16 weights)

  const dim3 blk(256);
  const dim3 gGemm(16, 16);
  const int DD = D_ * D_;

  // spatial GAT -> h (f32 + f16)
  gat_kernel<<<dim3(B_ * T_), blk, 0, stream>>>(
      X, V_Adap, Wl, bl, Wr, br, att, gb, Wfc, bfc, cls, hf32, hf16);

  for (int l = 0; l < 2; ++l) {
    const size_t wo = (size_t)l * DD;
    // Q, K, V projections
    cvt_f32_f16<<<2048, blk, 0, stream>>>(Wq + wo, wf, DD);
    gemm_wmma_kernel<<<gGemm, blk, 0, stream>>>(hf16, wf, bq + l * D_, nullptr, qf,
                                                B_ * T_, D_, D_, 0);
    cvt_f32_f16<<<2048, blk, 0, stream>>>(Wk + wo, wf, DD);
    gemm_wmma_kernel<<<gGemm, blk, 0, stream>>>(hf16, wf, bk + l * D_, nullptr, kf,
                                                B_ * T_, D_, D_, 0);
    cvt_f32_f16<<<2048, blk, 0, stream>>>(Wv + wo, wf, DD);
    gemm_wmma_kernel<<<gGemm, blk, 0, stream>>>(hf16, wf, bv + l * D_, nullptr, vf,
                                                B_ * T_, D_, D_, 0);
    // multi-head attention -> xf (f16)
    attn_kernel<<<dim3(B_ * NHEADS), blk, 0, stream>>>(qf, kf, vf, xf);
    // output projection -> tmpf (f32)
    cvt_f32_f16<<<2048, blk, 0, stream>>>(Wo + wo, wf, DD);
    gemm_wmma_kernel<<<gGemm, blk, 0, stream>>>(xf, wf, bo + l * D_, tmpf, nullptr,
                                                B_ * T_, D_, D_, 0);
    add_ln_kernel<<<dim3(B_ * T_), blk, 0, stream>>>(hf32, hf16, tmpf,
                                                     ln1g + l * D_, ln1b + l * D_);
    // FFN
    cvt_f32_f16<<<2048, blk, 0, stream>>>(W1 + wo, wf, DD);
    gemm_wmma_kernel<<<gGemm, blk, 0, stream>>>(hf16, wf, b1 + l * DFF, nullptr, xf,
                                                B_ * T_, DFF, D_, 1);
    cvt_f32_f16<<<2048, blk, 0, stream>>>(W2 + wo, wf, DD);
    gemm_wmma_kernel<<<gGemm, blk, 0, stream>>>(xf, wf, b2 + l * D_, tmpf, nullptr,
                                                B_ * T_, D_, DFF, 0);
    add_ln_kernel<<<dim3(B_ * T_), blk, 0, stream>>>(hf32, hf16, tmpf,
                                                     ln2g + l * D_, ln2b + l * D_);
  }

  final_kernel<<<dim3(2), blk, 0, stream>>>(hf32, Wout, bout, out);
}
